// GraphConvNet_20306605375975
// MI455X (gfx1250) — matrix operand order repaired and checked by reading, not compiled, not run on hardware
//
#include <hip/hip_runtime.h>
#include <hip/hip_bf16.h>
#include <math.h>

typedef __attribute__((ext_vector_type(16))) _Float16 v16h;
typedef __attribute__((ext_vector_type(8)))  float    v8f;

#define N_NODES 100000
#define N_EDGES 1600000
#define D_HID   128
#define D_OUT   64
#define KTILES  4                         // K=128 = 4 x 32
#define HID_FRAG_HALVES (8*KTILES*32*16)  // 16384 f16 per hidden layer
#define OUT_FRAG_BASE   (7*HID_FRAG_HALVES)

// ---------------- degree accumulation ----------------
__global__ void gcn_deg_kernel(const int* __restrict__ src, const int* __restrict__ dst,
                               float* __restrict__ degout, float* __restrict__ degin, int E) {
    int e = blockIdx.x * blockDim.x + threadIdx.x;
    if (e < E) {
        atomicAdd(&degout[src[e]], 1.0f);
        atomicAdd(&degin[dst[e]], 1.0f);
    }
}

// deg -> deg^{-1/2} with clamp-to-1 (DGL norm='both')
__global__ void gcn_norm_kernel(float* __restrict__ d, int n) {
    int i = blockIdx.x * blockDim.x + threadIdx.x;
    if (i < n) d[i] = rsqrtf(fmaxf(d[i], 1.0f));
}

// xs = features * norm_src (row-broadcast)
__global__ void gcn_scale_kernel(const float* __restrict__ f, const float* __restrict__ ns,
                                 float* __restrict__ xs, int total) {
    int i = blockIdx.x * blockDim.x + threadIdx.x;
    if (i < total) xs[i] = f[i] * ns[i >> 7];
}

// Pack W (f32 row-major [K, Ncols]) into WMMA f16 B-fragments:
// fragment(nt,kt): lane l holds column nt*16+l%16, element e at k = kt*32 + (e<8 ? 8h+e : 16+8h+e-8), h=l/16
__global__ void gcn_packw_kernel(const float* __restrict__ Wh, const float* __restrict__ Wo,
                                 _Float16* __restrict__ wp) {
    int t = blockIdx.x * blockDim.x + threadIdx.x;
    const int hidFrags = 7 * 8 * KTILES * 32;   // 7168
    const int outFrags = 4 * KTILES * 32;       // 512
    if (t >= hidFrags + outFrags) return;
    const float* W; int ncols; _Float16* dstp; int nt, kt, lane;
    if (t < hidFrags) {
        int L = t / (8 * KTILES * 32);
        int rem = t % (8 * KTILES * 32);
        nt = rem / (KTILES * 32); rem %= (KTILES * 32);
        kt = rem / 32; lane = rem % 32;
        W = Wh + (size_t)L * D_HID * D_HID; ncols = D_HID;
        dstp = wp + (size_t)L * HID_FRAG_HALVES + ((nt * KTILES + kt) * 32 + lane) * 16;
    } else {
        int r = t - hidFrags;
        nt = r / (KTILES * 32); int rem = r % (KTILES * 32);
        kt = rem / 32; lane = rem % 32;
        W = Wo; ncols = D_OUT;
        dstp = wp + OUT_FRAG_BASE + ((nt * KTILES + kt) * 32 + lane) * 16;
    }
    int h = lane >> 4;
    int n = nt * 16 + (lane & 15);
#pragma unroll
    for (int e = 0; e < 16; ++e) {
        int k = kt * 32 + (e < 8 ? 8 * h + e : 16 + 8 * h + (e - 8));
        dstp[e] = (_Float16)W[(size_t)k * ncols + n];
    }
}

// agg[dst] += xs[src] : 32 lanes/edge, 4 feats/lane. Edge indices are wave-uniform ->
// readfirstlane makes them SGPR so src/dst become scalar loads; gather stays float4 VMEM.
__global__ void gcn_spmm_kernel(const float* __restrict__ xs, const int* __restrict__ src,
                                const int* __restrict__ dst, float* __restrict__ agg, int E) {
    int wv   = __builtin_amdgcn_readfirstlane((int)threadIdx.x) >> 5;  // wave id in block
    int e    = blockIdx.x * 8 + wv;
    int lane = threadIdx.x & 31;
    if (e < E) {
        int s = src[e], d = dst[e];
        float4 v = *(const float4*)(xs + (size_t)s * D_HID + lane * 4);
        float* o = agg + (size_t)d * D_HID + lane * 4;
        atomicAdd(o + 0, v.x);
        atomicAdd(o + 1, v.y);
        atomicAdd(o + 2, v.z);
        atomicAdd(o + 3, v.w);
    }
}

// A fragment: 16x32 f16 k-tile; lane holds row lm, K = half*8 interleave (ISA 7.12.2)
__device__ inline v16h gcn_load_afrag(const float* __restrict__ rowbase, int kt, int half,
                                      float nd) {
    const float* p = rowbase + kt * 32 + half * 8;
    float4 q0 = *(const float4*)(p + 0);
    float4 q1 = *(const float4*)(p + 4);
    float4 q2 = *(const float4*)(p + 16);
    float4 q3 = *(const float4*)(p + 20);
    v16h a;
    a[0]  = (_Float16)(q0.x * nd); a[1]  = (_Float16)(q0.y * nd);
    a[2]  = (_Float16)(q0.z * nd); a[3]  = (_Float16)(q0.w * nd);
    a[4]  = (_Float16)(q1.x * nd); a[5]  = (_Float16)(q1.y * nd);
    a[6]  = (_Float16)(q1.z * nd); a[7]  = (_Float16)(q1.w * nd);
    a[8]  = (_Float16)(q2.x * nd); a[9]  = (_Float16)(q2.y * nd);
    a[10] = (_Float16)(q2.z * nd); a[11] = (_Float16)(q2.w * nd);
    a[12] = (_Float16)(q3.x * nd); a[13] = (_Float16)(q3.y * nd);
    a[14] = (_Float16)(q3.z * nd); a[15] = (_Float16)(q3.w * nd);
    return a;
}

// Fused: y = (agg * norm_dst) @ W + b ; hidden: xs_out = relu(y)*norm_src ; final: sigmoid(y).
// Block = 4 waves. W fragments async-copied to LDS once per block (ASYNCcnt path), then each
// wave computes 2 row-strips (32 rows) x NT col-tiles; B reused from LDS across both strips.
template <int NT, bool FINAL>
__global__ __launch_bounds__(128) void gcn_gemm_kernel(
        const float* __restrict__ agg, const float* __restrict__ norm_dst,
        const float* __restrict__ norm_src, const _Float16* __restrict__ wfrag,
        const float* __restrict__ bias, float* __restrict__ xs_out,
        float* __restrict__ final_out) {
    constexpr int FRAGS = NT * KTILES;                 // 32 (hidden) / 16 (final)
    __shared__ __align__(32) _Float16 bsh[FRAGS * 32 * 16];

    // ---- async stage of all B fragments (FRAGS KB) into LDS ----
    {
        uint32_t lbase = (uint32_t)(uintptr_t)&bsh[0];
        const char* gbase = (const char*)wfrag;
#pragma unroll
        for (int i = threadIdx.x; i < FRAGS * 64; i += 128) {   // 16B chunks, EXEC all-ones
            uint32_t loff = lbase + (uint32_t)i * 16u;
            uint64_t g = (uint64_t)(uintptr_t)(gbase + (size_t)i * 16u);
            asm volatile("global_load_async_to_lds_b128 %0, %1, off"
                         :: "v"(loff), "v"(g) : "memory");
        }
        asm volatile("s_wait_asynccnt 0x0" ::: "memory");
    }
    __syncthreads();

    int wave = blockIdx.x * 4 + ((int)threadIdx.x >> 5);
    if (wave >= N_NODES / 32) return;                  // wave-uniform; EXEC all-ones below
    int lane = threadIdx.x & 31;
    int half = lane >> 4, lm = lane & 15;
    int r0 = wave * 32;

    // A fragments for 2 row-strips, norm_dst folded into f32->f16 conversion
    v16h afrag[2][KTILES];
#pragma unroll
    for (int s = 0; s < 2; ++s) {
        int arow = r0 + s * 16 + lm;
        float nd = norm_dst[arow];
        const float* rowbase = agg + (size_t)arow * D_HID;
#pragma unroll
        for (int kt = 0; kt < KTILES; ++kt)
            afrag[s][kt] = gcn_load_afrag(rowbase, kt, half, nd);
    }

#pragma unroll
    for (int nt = 0; nt < NT; ++nt) {
        v16h bfrag[KTILES];
#pragma unroll
        for (int kt = 0; kt < KTILES; ++kt)
            bfrag[kt] = *(const v16h*)(bsh + ((nt * KTILES + kt) * 32 + lane) * 16);
        int col = nt * 16 + lm;
        float b = bias[col];
#pragma unroll
        for (int s = 0; s < 2; ++s) {
            v8f acc = {};
#pragma unroll
            for (int kt = 0; kt < KTILES; ++kt)
                acc = __builtin_amdgcn_wmma_f32_16x16x32_f16(
                    false, afrag[s][kt], false, bfrag[kt], (short)0, acc, false, false);
#pragma unroll
            for (int r = 0; r < 8; ++r) {              // C: row = r0+s*16+half*8+r, col = lm
                int rr = r0 + s * 16 + half * 8 + r;
                float y = acc[r] + b;
                if constexpr (FINAL) {
                    final_out[(size_t)rr * D_OUT + col] =
                        __builtin_amdgcn_rcpf(1.0f + __expf(-y));   // sigmoid via v_rcp_f32
                } else {
                    xs_out[(size_t)rr * D_HID + col] = fmaxf(y, 0.0f) * norm_src[rr];
                }
            }
        }
    }
}

extern "C" void kernel_launch(void* const* d_in, const int* in_sizes, int n_in,
                              void* d_out, int out_size, void* d_ws, size_t ws_size,
                              hipStream_t stream) {
    const float* features = (const float*)d_in[0];
    const float* W_hidden = (const float*)d_in[1];
    const float* b_hidden = (const float*)d_in[2];
    const float* W_out    = (const float*)d_in[3];
    const float* b_out    = (const float*)d_in[4];
    const int*   src      = (const int*)d_in[5];
    const int*   dst      = (const int*)d_in[6];

    // Workspace layout (~103.5 MB): xs | agg | norm_src | norm_dst | packed W (f16)
    float*    xs   = (float*)d_ws;
    float*    agg  = xs + (size_t)N_NODES * D_HID;
    float*    nsrc = agg + (size_t)N_NODES * D_HID;
    float*    ndst = nsrc + N_NODES;
    _Float16* wp   = (_Float16*)(ndst + N_NODES);   // byte offset is 32B-aligned

    // Degrees -> symmetric norms
    hipMemsetAsync(nsrc, 0, (size_t)2 * N_NODES * sizeof(float), stream);
    gcn_deg_kernel<<<(N_EDGES + 255) / 256, 256, 0, stream>>>(src, dst, nsrc, ndst, N_EDGES);
    gcn_norm_kernel<<<(2 * N_NODES + 255) / 256, 256, 0, stream>>>(nsrc, 2 * N_NODES);

    // Pack weights into WMMA f16 B-fragments (once per call)
    gcn_packw_kernel<<<(7680 + 255) / 256, 256, 0, stream>>>(W_hidden, W_out, wp);

    // xs = features * norm_src
    gcn_scale_kernel<<<(N_NODES * D_HID + 255) / 256, 256, 0, stream>>>(
        features, nsrc, xs, N_NODES * D_HID);

    const int spmmBlocks = (N_EDGES * 32) / 256;            // exact: 200000 (8 edges/block)
    const int gemmBlocks = (N_NODES / 32 + 3) / 4;          // 3125 waves -> 782 blocks

    for (int L = 0; L < 7; ++L) {
        hipMemsetAsync(agg, 0, (size_t)N_NODES * D_HID * sizeof(float), stream);
        gcn_spmm_kernel<<<spmmBlocks, 256, 0, stream>>>(xs, src, dst, agg, N_EDGES);
        gcn_gemm_kernel<8, false><<<gemmBlocks, 128, 0, stream>>>(
            agg, ndst, nsrc, wp + (size_t)L * HID_FRAG_HALVES,
            b_hidden + (size_t)L * D_HID, xs, nullptr);
    }

    // Output layer: (agg*norm_dst) @ W_out + b_out -> sigmoid
    hipMemsetAsync(agg, 0, (size_t)N_NODES * D_HID * sizeof(float), stream);
    gcn_spmm_kernel<<<spmmBlocks, 256, 0, stream>>>(xs, src, dst, agg, N_EDGES);
    gcn_gemm_kernel<4, true><<<gemmBlocks, 128, 0, stream>>>(
        agg, ndst, nsrc, wp + OUT_FRAG_BASE, b_out, nullptr, (float*)d_out);
}
